// ProteinFunctionModel_20667382628720
// MI455X (gfx1250) — matrix operand order repaired and checked by reading, not compiled, not run on hardware
//
#include <hip/hip_runtime.h>
#include <hip/hip_bf16.h>
#include <math.h>

#define D_EMB   2560
#define DM      512
#define NHEADS  8
#define HD      64
#define DFF     1024
#define NLAYERS 6
#define NGO     4273
#define NGOP    4288            // head N padded to multiple of 64 (and 8)
#define BSZ     4
#define LSEQ    1024
#define NTOK    1025
#define NP      1056            // padded token count (multiple of 32)
#define BNP     (BSZ * NP)      // 4224
#define BL      (BSZ * LSEQ)    // 4096

// CDNA5 async global->LDS copies (ISA 10.x / 15.18). Set to 0 if the
// assembler rejects the mnemonics; falls back to uint4 loads + LDS stores.
#define USE_ASYNC_ASM 1

typedef __attribute__((ext_vector_type(16))) __bf16 v16bf;
typedef __attribute__((ext_vector_type(8)))  float  v8f;

__device__ __forceinline__ unsigned short f2bf(float f) {
    union { float f; unsigned int u; } c; c.f = f;
    unsigned int u = c.u;
    u += 0x7FFFu + ((u >> 16) & 1u);        // round-to-nearest-even
    return (unsigned short)(u >> 16);
}

union FragBF { v16bf v; unsigned short u[16]; };

// ---------------------------------------------------------------------------
// Generic bf16 WMMA GEMM:  C = epilogue( alpha * A[M,K] @ B[K,N] )
//   A: bf16 row-major, lda (multiple of 8, 16B-aligned base)
//   B: bf16; bT==0 -> B[k*ldb+n] (N region padded to block tile),
//            bT==1 -> B[n*ldb+k], rows guarded by nB
//   epilogue: v = alpha*AB + row_bias[n]; if act==1 v = gelu(v); v += add_mat[m,n]
// Block = 256 threads (8 waves), 128x64 tile, each wave 32x32 (2x2 WMMA),
// K stepped by 32. A (and B when bT=1) staged with async global->LDS b128,
// double-buffered in LDS so the DMA of tile k+1 overlaps the WMMAs of tile k.
// ---------------------------------------------------------------------------
__global__ __launch_bounds__(256)
void gemm_bf16_kernel(const unsigned short* __restrict__ A, int lda,
                      const unsigned short* __restrict__ Bm, int ldb, int bT, int nB,
                      void* __restrict__ C, int ldc, int out_bf16,
                      int M, int N, int K, float alpha,
                      const float* __restrict__ row_bias,
                      const float* __restrict__ add_mat, int ld_add,
                      int act)
{
    __shared__ __align__(16) unsigned short As[2][128][40];
    __shared__ __align__(16) unsigned short Bs[2][64][40];

    const int tid  = threadIdx.x;
    const int m0   = blockIdx.y * 128;
    const int n0   = blockIdx.x * 64;
    const int lane = tid & 31;
    const int wave = tid >> 5;
    const int wr   = (wave >> 1) * 32;   // wave row offset in tile (0..96)
    const int wc   = (wave & 1) * 32;    // wave col offset in tile (0/32)
    const int lm   = lane & 15;
    const int lh   = lane >> 4;          // lane half (0/1)

    // stage one 128x32 A tile + 64x32 B tile (as Bs[n][k]) for K-offset kk
    auto stage = [&](int buf, int kk) {
        // ---- A tile: 512 chunks of 16B, 2 per thread ----
#pragma unroll
        for (int s = 0; s < 2; ++s) {
            int t  = tid + s * 256;
            int r  = t >> 2, cc = t & 3;
            int gm = m0 + r;
            const unsigned short* gp = A + (long)gm * lda + kk + cc * 8;
#if USE_ASYNC_ASM
            unsigned lo = (unsigned)(unsigned long long)&As[buf][r][cc * 8];
            if (gm < M)
                asm volatile("global_load_async_to_lds_b128 %0, %1, off"
                             :: "v"(lo), "v"((unsigned long long)gp) : "memory");
#else
            uint4 v = {0u, 0u, 0u, 0u};
            if (gm < M) v = *(const uint4*)gp;
            *(uint4*)&As[buf][r][cc * 8] = v;
#endif
        }
        // ---- B tile ----
        if (bT) {
            // B[n*ldb + k]: contiguous in k -> 64x4 chunks, 1 per thread
            int n  = tid >> 2, cc = tid & 3;
            int gn = n0 + n;
            const unsigned short* gp = Bm + (long)gn * ldb + kk + cc * 8;
#if USE_ASYNC_ASM
            unsigned lo = (unsigned)(unsigned long long)&Bs[buf][n][cc * 8];
            if (gn < nB)
                asm volatile("global_load_async_to_lds_b128 %0, %1, off"
                             :: "v"(lo), "v"((unsigned long long)gp) : "memory");
#else
            uint4 v = {0u, 0u, 0u, 0u};
            if (gn < nB) v = *(const uint4*)gp;
            *(uint4*)&Bs[buf][n][cc * 8] = v;
#endif
        } else if (tid < 128) {
            // B[k*ldb + n]: load 2 adjacent k-rows of 8 n, transpose into LDS
            int k  = (tid >> 3) * 2;
            int nn = tid & 7;
            union { uint4 v; unsigned short h[8]; } ra, rb;
            ra.v = *(const uint4*)(Bm + (long)(kk + k)     * ldb + n0 + nn * 8);
            rb.v = *(const uint4*)(Bm + (long)(kk + k + 1) * ldb + n0 + nn * 8);
#pragma unroll
            for (int e = 0; e < 8; ++e) {
                unsigned pack = (unsigned)ra.h[e] | ((unsigned)rb.h[e] << 16);
                *(unsigned*)&Bs[buf][nn * 8 + e][k] = pack; // k even -> dword aligned
            }
        }
    };

    v8f acc[2][2];
#pragma unroll
    for (int i = 0; i < 2; ++i)
#pragma unroll
        for (int j = 0; j < 2; ++j)
#pragma unroll
            for (int e = 0; e < 8; ++e) acc[i][j][e] = 0.0f;

    // prologue: fill buffer 0
    stage(0, 0);
#if USE_ASYNC_ASM
    asm volatile("s_wait_asynccnt 0x0" ::: "memory");
#endif
    __syncthreads();

    for (int k0 = 0; k0 < K; k0 += 32) {
        const int buf = (k0 >> 5) & 1;
        // kick off DMA of the next tile into the other buffer
        if (k0 + 32 < K) stage(buf ^ 1, k0 + 32);

        // ---- fragment loads (contiguous b128 reads per CDNA5 VGPR layouts) ----
        FragBF a[2], b[2];
#pragma unroll
        for (int i = 0; i < 2; ++i) {
            int row = wr + i * 16 + lm;
            int kb  = lh * 8;
#pragma unroll
            for (int e = 0; e < 8; ++e) {
                a[i].u[e]     = As[buf][row][kb + e];      // VGPR0..3: k=kb..kb+7
                a[i].u[e + 8] = As[buf][row][kb + 16 + e]; // VGPR4..7: k=kb+16..23
            }
        }
#pragma unroll
        for (int j = 0; j < 2; ++j) {
            int n  = wc + j * 16 + lm;
            int kb = lh * 8;
#pragma unroll
            for (int e = 0; e < 16; ++e) b[j].u[e] = Bs[buf][n][kb + e + (e >> 3) * 8];
        }

#pragma unroll
        for (int i = 0; i < 2; ++i)
#pragma unroll
            for (int j = 0; j < 2; ++j)
                acc[i][j] = __builtin_amdgcn_wmma_f32_16x16x32_bf16(
                    false, a[i].v, false, b[j].v, (short)0, acc[i][j], false, false);

#if USE_ASYNC_ASM
        asm volatile("s_wait_asynccnt 0x0" ::: "memory");
#endif
        __syncthreads();   // all waves: done reading buf, done writing buf^1
    }

    // epilogue; C layout: VGPR r -> M = base + lh*8 + r, N = base + lm
#pragma unroll
    for (int i = 0; i < 2; ++i)
#pragma unroll
        for (int j = 0; j < 2; ++j) {
            int mb = m0 + wr + i * 16 + lh * 8;
            int nb = n0 + wc + j * 16 + lm;
            if (nb < N) {
                float rb = row_bias ? row_bias[nb] : 0.0f;
#pragma unroll
                for (int r = 0; r < 8; ++r) {
                    int gm = mb + r;
                    if (gm < M) {
                        float v = alpha * acc[i][j][r] + rb;
                        if (act == 1) v = 0.5f * v * (1.0f + erff(v * 0.70710678118f));
                        if (add_mat) v += add_mat[(long)gm * ld_add + nb];
                        if (out_bf16)
                            ((unsigned short*)C)[(long)gm * ldc + nb] = f2bf(v);
                        else
                            ((float*)C)[(long)gm * ldc + nb] = v;
                    }
                }
            }
        }
}

// ---------------------------------------------------------------------------
__global__ void f32_to_bf16_kernel(const float* __restrict__ in,
                                   unsigned short* __restrict__ out, long n) {
    long i = (long)blockIdx.x * blockDim.x + threadIdx.x;
    if (i < n) out[i] = f2bf(in[i]);
}

// convert with column padding: dst[r, 0:scols) = bf16(src), dst[r, scols:dcols) = 0
__global__ void convert_pad_kernel(const float* __restrict__ in,
                                   unsigned short* __restrict__ out,
                                   int scols, int dcols, long n) {
    long i = (long)blockIdx.x * blockDim.x + threadIdx.x;
    if (i >= n) return;
    long r = i / dcols;
    int  c = (int)(i % dcols);
    out[i] = (c < scols) ? f2bf(in[r * scols + c]) : (unsigned short)0;
}

// small per-token projections (K<=3) written as bf16 into feature matrix cols 512..2559
__global__ void small_feats_kernel(const float* __restrict__ plddt,
                                   const float* __restrict__ cent,
                                   const float* __restrict__ orient,
                                   const float* __restrict__ sc,
                                   const float* __restrict__ plddt_w, const float* __restrict__ plddt_b,
                                   const float* __restrict__ cent_w,  const float* __restrict__ cent_b,
                                   const float* __restrict__ orient_w,const float* __restrict__ orient_b,
                                   const float* __restrict__ sc_w,    const float* __restrict__ sc_b,
                                   unsigned short* __restrict__ F) {
    int t = blockIdx.x;                       // token 0..BL-1
    int d = blockIdx.y * 256 + threadIdx.x;   // 0..511
    float p  = plddt[t];
    float c0 = cent[t*3], c1 = cent[t*3+1], c2 = cent[t*3+2];
    float o0 = orient[t*3], o1 = orient[t*3+1], o2 = orient[t*3+2];
    float s0 = sc[t*3], s1 = sc[t*3+1], s2 = sc[t*3+2];
    long base = (long)t * D_EMB;
    F[base +  512 + d] = f2bf(p * plddt_w[d] + plddt_b[d]);
    F[base + 1024 + d] = f2bf(c0*cent_w[d]   + c1*cent_w[512+d]   + c2*cent_w[1024+d]   + cent_b[d]);
    F[base + 1536 + d] = f2bf(o0*orient_w[d] + o1*orient_w[512+d] + o2*orient_w[1024+d] + orient_b[d]);
    F[base + 2048 + d] = f2bf(s0*sc_w[d]     + s1*sc_w[512+d]     + s2*sc_w[1024+d]     + sc_b[d]);
}

// build x[B,NP,DM] fp32: row0 = cls, rows 1..L = fused + pos_enc, rest 0
__global__ void assemble_kernel(const float* __restrict__ xtmp,
                                const float* __restrict__ pos_enc,
                                const float* __restrict__ cls,
                                float* __restrict__ x) {
    int row = blockIdx.x;
    int d   = blockIdx.y * 256 + threadIdx.x;
    int b = row / NP, r = row % NP;
    float v;
    if (r == 0)            v = cls[d];
    else if (r <= LSEQ)    v = xtmp[((long)(b*LSEQ + r - 1)) * DM + d] + pos_enc[(long)(r-1)*DM + d];
    else                   v = 0.0f;
    x[(long)row * DM + d] = v;
}

// bias[b,i,j] = pad(pairwise scalar) + relative-position bucket bias; -1e30 on pad
__global__ void pair_bias_kernel(const float* __restrict__ eucl,
                                 const float* __restrict__ pae,
                                 const float* __restrict__ ev,
                                 const float* __restrict__ pair_w,
                                 const float* __restrict__ pair_b,
                                 const float* __restrict__ rel_emb,
                                 float* __restrict__ bias) {
    long idx = (long)blockIdx.x * blockDim.x + threadIdx.x;
    long per_b = (long)NP * NP;
    if (idx >= per_b) return;
    int b = blockIdx.y;
    int i = (int)(idx / NP), j = (int)(idx % NP);
    float v;
    if (j >= NTOK || i >= NTOK) {
        v = -1e30f;
    } else {
        int rel = j - i;
        int dist = rel < 0 ? -rel : rel;
        int bucket;
        if (dist < 16) bucket = dist;
        else {
            float large = fminf(fmaxf((float)dist, 16.0f), 128.0f);
            bucket = (int)(logf(large / 16.0f + 1e-8f) / logf(8.0f + 1e-8f) * 15.0f) + 16;
        }
        if (rel < 0) bucket += 16;
        bucket = bucket < 0 ? 0 : (bucket > 31 ? 31 : bucket);
        v = rel_emb[bucket];
        if (i >= 1 && j >= 1) {
            long o2 = ((long)b * LSEQ + (i - 1)) * LSEQ + (j - 1);
            v += eucl[o2]*pair_w[0] + pae[o2]*pair_w[1]
               + ev[o2*3]*pair_w[2] + ev[o2*3+1]*pair_w[3] + ev[o2*3+2]*pair_w[4]
               + pair_b[0];
        }
    }
    bias[(long)b * per_b + idx] = v;
}

__global__ __launch_bounds__(256)
void layernorm_kernel(const float* __restrict__ x,
                      const float* __restrict__ g, const float* __restrict__ bb,
                      unsigned short* __restrict__ z) {
    __shared__ float red[256];
    long row = blockIdx.x;
    const float* xr = x + row * DM;
    int tid = threadIdx.x;
    float v0 = xr[tid], v1 = xr[tid + 256];
    red[tid] = v0 + v1;
    __syncthreads();
    for (int o = 128; o > 0; o >>= 1) { if (tid < o) red[tid] += red[tid + o]; __syncthreads(); }
    float mean = red[0] * (1.0f / DM);
    __syncthreads();
    float d0 = v0 - mean, d1 = v1 - mean;
    red[tid] = d0 * d0 + d1 * d1;
    __syncthreads();
    for (int o = 128; o > 0; o >>= 1) { if (tid < o) red[tid] += red[tid + o]; __syncthreads(); }
    float rstd = rsqrtf(red[0] * (1.0f / DM) + 1e-5f);
    z[row * DM + tid]       = f2bf(d0 * rstd * g[tid]       + bb[tid]);
    z[row * DM + tid + 256] = f2bf(d1 * rstd * g[tid + 256] + bb[tid + 256]);
}

__global__ __launch_bounds__(256)
void softmax_kernel(const float* __restrict__ S, unsigned short* __restrict__ P) {
    __shared__ float red[256];
    const float* sr = S + (long)blockIdx.x * NP;
    unsigned short* pr = P + (long)blockIdx.x * NP;
    int tid = threadIdx.x;
    float mx = -1e38f;
    for (int c = tid; c < NP; c += 256) mx = fmaxf(mx, sr[c]);
    red[tid] = mx; __syncthreads();
    for (int o = 128; o > 0; o >>= 1) { if (tid < o) red[tid] = fmaxf(red[tid], red[tid + o]); __syncthreads(); }
    mx = red[0]; __syncthreads();
    float sum = 0.0f;
    for (int c = tid; c < NP; c += 256) sum += expf(sr[c] - mx);
    red[tid] = sum; __syncthreads();
    for (int o = 128; o > 0; o >>= 1) { if (tid < o) red[tid] += red[tid + o]; __syncthreads(); }
    float inv = 1.0f / red[0];
    for (int c = tid; c < NP; c += 256) pr[c] = f2bf(expf(sr[c] - mx) * inv);
}

__global__ void gather_cls_kernel(const unsigned short* __restrict__ Z,
                                  unsigned short* __restrict__ out) {
    int b = blockIdx.x, d = threadIdx.x;     // block = 512 threads
    out[b * DM + d] = Z[(long)b * NP * DM + d];
}

// ---------------------------------------------------------------------------
extern "C" void kernel_launch(void* const* d_in, const int* in_sizes, int n_in,
                              void* d_out, int out_size, void* d_ws, size_t ws_size,
                              hipStream_t stream) {
    (void)in_sizes; (void)n_in; (void)out_size; (void)ws_size;
    const float* seq_embed   = (const float*)d_in[0];
    const float* pae         = (const float*)d_in[1];
    const float* plddt       = (const float*)d_in[2];
    const float* centroid    = (const float*)d_in[3];
    const float* orient_v    = (const float*)d_in[4];
    const float* sc_v        = (const float*)d_in[5];
    const float* eucl        = (const float*)d_in[6];
    const float* edge_v      = (const float*)d_in[7];
    const float* input_proj_w= (const float*)d_in[8];
    const float* input_proj_b= (const float*)d_in[9];
    const float* plddt_w     = (const float*)d_in[10];
    const float* plddt_b     = (const float*)d_in[11];
    const float* cent_w      = (const float*)d_in[12];
    const float* cent_b      = (const float*)d_in[13];
    const float* orient_w    = (const float*)d_in[14];
    const float* orient_b    = (const float*)d_in[15];
    const float* sc_w        = (const float*)d_in[16];
    const float* sc_b        = (const float*)d_in[17];
    const float* fusion_w    = (const float*)d_in[18];
    const float* fusion_b    = (const float*)d_in[19];
    const float* pair_w      = (const float*)d_in[20];
    const float* pair_b      = (const float*)d_in[21];
    const float* pos_enc     = (const float*)d_in[22];
    const float* cls_token   = (const float*)d_in[23];
    const float* rel_emb     = (const float*)d_in[24];
    const float* ln1_g       = (const float*)d_in[25];
    const float* ln1_b       = (const float*)d_in[26];
    const float* ln2_g       = (const float*)d_in[27];
    const float* ln2_b       = (const float*)d_in[28];
    const float* in_proj_w   = (const float*)d_in[29];
    const float* in_proj_b   = (const float*)d_in[30];
    const float* out_proj_w  = (const float*)d_in[31];
    const float* out_proj_b  = (const float*)d_in[32];
    const float* ff1_w       = (const float*)d_in[33];
    const float* ff1_b       = (const float*)d_in[34];
    const float* ff2_w       = (const float*)d_in[35];
    const float* ff2_b       = (const float*)d_in[36];
    const float* norm_g      = (const float*)d_in[37];
    const float* norm_b      = (const float*)d_in[38];
    const float* head_w      = (const float*)d_in[39];
    const float* head_b      = (const float*)d_in[40];

    char* ws = (char*)d_ws;
    size_t off = 0;
    auto carve = [&](size_t bytes) -> char* {
        char* p = ws + off;
        off = (off + bytes + 255) & ~(size_t)255;
        return p;
    };
    unsigned short* seqbf  = (unsigned short*)carve((size_t)BL * D_EMB * 2);
    unsigned short* Fbuf   = (unsigned short*)carve((size_t)BL * D_EMB * 2);
    unsigned short* w_in   = (unsigned short*)carve((size_t)D_EMB * DM * 2);
    unsigned short* w_fus  = (unsigned short*)carve((size_t)D_EMB * DM * 2);
    unsigned short* w_qkv  = (unsigned short*)carve((size_t)NLAYERS * DM * (3*DM) * 2);
    unsigned short* w_out  = (unsigned short*)carve((size_t)NLAYERS * DM * DM * 2);
    unsigned short* w_ff1  = (unsigned short*)carve((size_t)NLAYERS * DM * DFF * 2);
    unsigned short* w_ff2  = (unsigned short*)carve((size_t)NLAYERS * DFF * DM * 2);
    unsigned short* w_head = (unsigned short*)carve((size_t)DM * NGOP * 2);
    float*          xtmp   = (float*)carve((size_t)BL * DM * 4);
    float*          x      = (float*)carve((size_t)BNP * DM * 4);
    unsigned short* Z      = (unsigned short*)carve((size_t)BNP * DM * 2);
    unsigned short* QKV    = (unsigned short*)carve((size_t)BNP * (3*DM) * 2);
    float*          biasb  = (float*)carve((size_t)BSZ * NP * NP * 4);
    float*          Sbuf   = (float*)carve((size_t)NHEADS * NP * NP * 4);
    unsigned short* Pbuf   = (unsigned short*)carve((size_t)NHEADS * NP * NP * 2);
    unsigned short* Obf    = (unsigned short*)carve((size_t)BNP * DM * 2);
    unsigned short* Hbuf   = (unsigned short*)carve((size_t)BNP * DFF * 2);
    unsigned short* Zcls   = (unsigned short*)carve((size_t)BSZ * DM * 2);

    auto conv = [&](const float* src, unsigned short* dst, long n) {
        f32_to_bf16_kernel<<<dim3((unsigned)((n + 255) / 256)), dim3(256), 0, stream>>>(src, dst, n);
    };
    auto gemm = [&](const unsigned short* A, int lda,
                    const unsigned short* Bm, int ldb, int bT, int nB,
                    void* C, int ldc, int obf, int M, int N, int K, float alpha,
                    const float* rowb, const float* addm, int ldadd, int act) {
        dim3 g((N + 63) / 64, (M + 127) / 128);
        gemm_bf16_kernel<<<g, dim3(256), 0, stream>>>(A, lda, Bm, ldb, bT, nB,
                                                      C, ldc, obf, M, N, K, alpha,
                                                      rowb, addm, ldadd, act);
    };

    // --- weight / activation conversions to bf16 ---
    conv(seq_embed,    seqbf,  (long)BL * D_EMB);
    conv(input_proj_w, w_in,   (long)D_EMB * DM);
    conv(fusion_w,     w_fus,  (long)D_EMB * DM);
    conv(in_proj_w,    w_qkv,  (long)NLAYERS * DM * (3*DM));
    conv(out_proj_w,   w_out,  (long)NLAYERS * DM * DM);
    conv(ff1_w,        w_ff1,  (long)NLAYERS * DM * DFF);
    conv(ff2_w,        w_ff2,  (long)NLAYERS * DFF * DM);
    {
        long n = (long)DM * NGOP;
        convert_pad_kernel<<<dim3((unsigned)((n + 255) / 256)), dim3(256), 0, stream>>>(
            head_w, w_head, NGO, NGOP, n);
    }

    // --- input fusion ---
    // F[:, 0:512] = seq_embed @ input_proj_w + b   (bf16 out, ldc = D_EMB)
    gemm(seqbf, D_EMB, w_in, DM, 0, DM, Fbuf, D_EMB, 1, BL, DM, D_EMB, 1.0f,
         input_proj_b, nullptr, 0, 0);
    small_feats_kernel<<<dim3(BL, 2), dim3(256), 0, stream>>>(
        plddt, centroid, orient_v, sc_v,
        plddt_w, plddt_b, cent_w, cent_b, orient_w, orient_b, sc_w, sc_b, Fbuf);
    // xtmp = F @ fusion_w + b (fp32)
    gemm(Fbuf, D_EMB, w_fus, DM, 0, DM, xtmp, DM, 0, BL, DM, D_EMB, 1.0f,
         fusion_b, nullptr, 0, 0);
    assemble_kernel<<<dim3(BNP, 2), dim3(256), 0, stream>>>(xtmp, pos_enc, cls_token, x);
    pair_bias_kernel<<<dim3((unsigned)(((long)NP * NP + 255) / 256), BSZ), dim3(256), 0, stream>>>(
        eucl, pae, edge_v, pair_w, pair_b, rel_emb, biasb);

    const float scale = 1.0f / 8.0f;   // 1/sqrt(64)
    for (int l = 0; l < NLAYERS; ++l) {
        layernorm_kernel<<<dim3(BNP), dim3(256), 0, stream>>>(
            x, ln1_g + l * DM, ln1_b + l * DM, Z);
        // QKV = Z @ Wqkv + b (bf16)
        gemm(Z, DM, w_qkv + (size_t)l * DM * (3*DM), 3*DM, 0, 3*DM,
             QKV, 3*DM, 1, BNP, 3*DM, DM, 1.0f, in_proj_b + l * (3*DM), nullptr, 0, 0);

        for (int b = 0; b < BSZ; ++b) {
            const unsigned short* qkv_b = QKV + (size_t)b * NP * (3*DM);
            for (int h = 0; h < NHEADS; ++h) {
                // S[h] = scale * Q Kt + bias[b]   (fp32)
                gemm(qkv_b + h * HD, 3*DM,
                     qkv_b + DM + h * HD, 3*DM, 1, NP,
                     Sbuf + (size_t)h * NP * NP, NP, 0,
                     NP, NP, HD, scale,
                     nullptr, biasb + (size_t)b * NP * NP, NP, 0);
            }
            softmax_kernel<<<dim3(NHEADS * NP), dim3(256), 0, stream>>>(Sbuf, Pbuf);
            for (int h = 0; h < NHEADS; ++h) {
                // O[:, h*64:...] = P[h] @ V_h  (bf16)
                gemm(Pbuf + (size_t)h * NP * NP, NP,
                     qkv_b + 2*DM + h * HD, 3*DM, 0, HD,
                     Obf + (size_t)b * NP * DM + h * HD, DM, 1,
                     NP, HD, NP, 1.0f, nullptr, nullptr, 0, 0);
            }
        }
        // x += O @ Wout + b
        gemm(Obf, DM, w_out + (size_t)l * DM * DM, DM, 0, DM,
             x, DM, 0, BNP, DM, DM, 1.0f, out_proj_b + l * DM, x, DM, 0);

        layernorm_kernel<<<dim3(BNP), dim3(256), 0, stream>>>(
            x, ln2_g + l * DM, ln2_b + l * DM, Z);
        // H = gelu(Z @ W1 + b)  (bf16)
        gemm(Z, DM, w_ff1 + (size_t)l * DM * DFF, DFF, 0, DFF,
             Hbuf, DFF, 1, BNP, DFF, DM, 1.0f, ff1_b + l * DFF, nullptr, 0, 1);
        // x += H @ W2 + b
        gemm(Hbuf, DFF, w_ff2 + (size_t)l * DFF * DM, DM, 0, DM,
             x, DM, 0, BNP, DM, DFF, 1.0f, ff2_b + l * DM, x, DM, 0);
    }

    layernorm_kernel<<<dim3(BNP), dim3(256), 0, stream>>>(x, norm_g, norm_b, Z);
    gather_cls_kernel<<<dim3(BSZ), dim3(512), 0, stream>>>(Z, Zcls);
    // out = Zcls @ head_w + head_b  (fp32, 4 x 4273)
    gemm(Zcls, DM, w_head, NGOP, 0, NGOP, d_out, NGO, 0, BSZ, NGO, DM, 1.0f,
         head_b, nullptr, 0, 0);
}
